// BatchRelationalWithoutLocationsModule_10917806867211
// MI455X (gfx1250) — compile-verified, matrix operand order checked by reading, not run on hardware
//
#include <hip/hip_runtime.h>
#include <hip/hip_bf16.h>

typedef __attribute__((ext_vector_type(16))) __bf16 v16bf;
typedef __attribute__((ext_vector_type(8)))  float  v8f;

constexpr int kB   = 32;   // batch
constexpr int kC   = 64;   // channels
constexpr int kL   = 256;  // h*w
constexpr int kHid = 32;   // hidden

// ---------------------------------------------------------------------------
// Kernel 1: A'[b,l,:] = b0 + x[b,:,l] @ W0[:64]; Bm[b,l,:] = x[b,:,l] @ W0[64:]
// ---------------------------------------------------------------------------
__global__ __launch_bounds__(256) void k_embed(const float* __restrict__ x,
                                               const float* __restrict__ W0,
                                               const float* __restrict__ b0,
                                               float* __restrict__ Ap,
                                               float* __restrict__ Bm) {
  int idx = blockIdx.x * 256 + threadIdx.x;   // [0, 32*256*32)
  int n = idx & (kHid - 1);
  int l = (idx >> 5) & (kL - 1);
  int b = idx >> 13;
  const float* xr = x + (size_t)(b * kC) * kL + l;
  float accA = 0.f, accB = 0.f;
#pragma unroll 4
  for (int ch = 0; ch < kC; ++ch) {
    float xv = xr[ch * kL];
    accA += xv * W0[ch * kHid + n];
    accB += xv * W0[(kC + ch) * kHid + n];
  }
  Ap[idx] = accA + b0[n];
  Bm[idx] = accB;
}

__global__ void k_zero(float* __restrict__ S) { S[threadIdx.x] = 0.f; }

// ---------------------------------------------------------------------------
// Kernel 2: fused pair MLP + reduction, WMMA bf16 path
// ---------------------------------------------------------------------------
__device__ __forceinline__ v8f wmma_bf16(v16bf a, v16bf b, v8f c) {
  return __builtin_amdgcn_wmma_f32_16x16x32_bf16(false, a, false, b,
                                                 (short)0, c, false, false);
}

__global__ __launch_bounds__(256) void k_pairs(const float* __restrict__ Ap,
                                               const float* __restrict__ Bm,
                                               const float* __restrict__ W1,
                                               const float* __restrict__ b1,
                                               const float* __restrict__ W2,
                                               const float* __restrict__ b2,
                                               float* __restrict__ S) {
  __shared__ float lA[kL * kHid];          // 32 KB: A' for this batch
  __shared__ float lB[16 * kHid];          // 2 KB:  B rows for this i-group
  __shared__ float trans[8][16 * kHid];    // 16 KB: per-wave C->A transpose scratch

  const int b    = blockIdx.x >> 4;
  const int igrp = blockIdx.x & 15;
  const int wave = threadIdx.x >> 5;
  const int lane = threadIdx.x & 31;
  const int lh   = lane >> 4;              // lane half: selects K sub-blocks
  const int lr   = lane & 15;
  const int kb0  = 8 * lh;                 // K block 0 base for this half
  const int kb1  = 16 + 8 * lh;            // K block 1 base

  // --- stage A' (whole batch) and B (16 rows) into LDS ---
  const float* ApB = Ap + (size_t)b * (kL * kHid);
  for (int i = threadIdx.x; i < kL * kHid; i += 256) lA[i] = ApB[i];
  const float* BmB = Bm + (size_t)(b * kL + igrp * 16) * kHid;
  for (int i = threadIdx.x; i < 16 * kHid; i += 256) lB[i] = BmB[i];
  __syncthreads();

  // --- preload W1/W2 in the bf16 B-operand layout (column n = 16*t + lr) ---
  v16bf w1b0, w1b1, w2b0, w2b1;
#pragma unroll
  for (int e = 0; e < 8; ++e) {
    w1b0[e]     = (__bf16)W1[(kb0 + e) * kHid + lr];
    w1b0[8 + e] = (__bf16)W1[(kb1 + e) * kHid + lr];
    w1b1[e]     = (__bf16)W1[(kb0 + e) * kHid + 16 + lr];
    w1b1[8 + e] = (__bf16)W1[(kb1 + e) * kHid + 16 + lr];
    w2b0[e]     = (__bf16)W2[(kb0 + e) * kHid + lr];
    w2b0[8 + e] = (__bf16)W2[(kb1 + e) * kHid + lr];
    w2b1[e]     = (__bf16)W2[(kb0 + e) * kHid + 16 + lr];
    w2b1[8 + e] = (__bf16)W2[(kb1 + e) * kHid + 16 + lr];
  }
  const float b1v0 = b1[lr], b1v1 = b1[16 + lr];
  const float b2v0 = b2[lr], b2v1 = b2[16 + lr];

  float acc0 = 0.f, acc1 = 0.f;
  float* tr = trans[wave];

  // 256 tiles per block (16 i's x 16 j-tiles), 32 per wave
  for (int task = wave; task < 256; task += 8) {
    const int ii = task >> 4;              // i row within group
    const int j0 = (task & 15) << 4;       // j tile base

    // h1 tile (16 rows x 32) built directly in A-operand layout:
    // lane row = lr, K columns {kb0..kb0+7} U {kb1..kb1+7}
    const float* arow = &lA[(j0 + lr) * kHid];
    const float* brow = &lB[ii * kHid];
    v16bf a1;
#pragma unroll
    for (int e = 0; e < 8; ++e) {
      a1[e]     = (__bf16)fmaxf(arow[kb0 + e] + brow[kb0 + e], 0.f);
      a1[8 + e] = (__bf16)fmaxf(arow[kb1 + e] + brow[kb1 + e], 0.f);
    }

    v8f z = {};
    v8f c0 = wmma_bf16(a1, w1b0, z);       // h2 cols 0..15  (C layout)
    v8f c1 = wmma_bf16(a1, w1b1, z);       // h2 cols 16..31

    // bias + relu, park in LDS as plain 16x32 f32 (C layout -> row major)
#pragma unroll
    for (int r = 0; r < 8; ++r) {
      tr[(r + 8 * lh) * kHid + lr]      = fmaxf(c0[r] + b1v0, 0.f);
      tr[(r + 8 * lh) * kHid + 16 + lr] = fmaxf(c1[r] + b1v1, 0.f);
    }
    // same-wave DS ops are in-order: loads below see stores above

    // reload as A-operand bf16 (transpose of lane<->vgpr mapping)
    const float* trow = &tr[lr * kHid];
    v16bf a2;
#pragma unroll
    for (int e = 0; e < 8; ++e) {
      a2[e]     = (__bf16)trow[kb0 + e];
      a2[8 + e] = (__bf16)trow[kb1 + e];
    }

    v8f d0 = wmma_bf16(a2, w2b0, z);
    v8f d1 = wmma_bf16(a2, w2b1, z);

    // h3 bias + relu + sum over the tile's 16 rows (per-lane partial)
#pragma unroll
    for (int r = 0; r < 8; ++r) {
      acc0 += fmaxf(d0[r] + b2v0, 0.f);
      acc1 += fmaxf(d1[r] + b2v1, 0.f);
    }
  }

  // fold lane halves (same N, complementary M), then accumulate globally
  acc0 += __shfl_xor(acc0, 16, 32);
  acc1 += __shfl_xor(acc1, 16, 32);
  if (lane < 16) {
    atomicAdd(&S[b * kHid + lr], acc0);
    atomicAdd(&S[b * kHid + 16 + lr], acc1);
  }
}

// ---------------------------------------------------------------------------
// Kernel 3: head  out = relu(S@Wp+bp) @ Wo + bo   (tiny, one block)
// ---------------------------------------------------------------------------
__global__ __launch_bounds__(1024) void k_head(const float* __restrict__ S,
                                               const float* __restrict__ Wp,
                                               const float* __restrict__ bp,
                                               const float* __restrict__ Wo,
                                               const float* __restrict__ bo,
                                               float* __restrict__ out) {
  __shared__ float fsh[kB][kHid];
  const int n = threadIdx.x, bb = threadIdx.y;
  float acc = bp[n];
#pragma unroll
  for (int k = 0; k < kHid; ++k) acc += S[bb * kHid + k] * Wp[k * kHid + n];
  fsh[bb][n] = fmaxf(acc, 0.f);
  __syncthreads();
  float o = bo[n];
#pragma unroll
  for (int k = 0; k < kHid; ++k) o += fsh[bb][k] * Wo[k * kHid + n];
  out[bb * kHid + n] = o;
}

// ---------------------------------------------------------------------------
extern "C" void kernel_launch(void* const* d_in, const int* in_sizes, int n_in,
                              void* d_out, int out_size, void* d_ws, size_t ws_size,
                              hipStream_t stream) {
  const float* x  = (const float*)d_in[0];
  const float* W0 = (const float*)d_in[1];
  const float* b0 = (const float*)d_in[2];
  const float* W1 = (const float*)d_in[3];
  const float* b1 = (const float*)d_in[4];
  const float* W2 = (const float*)d_in[5];
  const float* b2 = (const float*)d_in[6];
  const float* Wp = (const float*)d_in[7];
  const float* bp = (const float*)d_in[8];
  const float* Wo = (const float*)d_in[9];
  const float* bo = (const float*)d_in[10];

  float* ws = (float*)d_ws;
  float* Ap = ws;                           // [32][256][32] f32 (1 MB)
  float* Bm = ws + kB * kL * kHid;          // [32][256][32] f32 (1 MB)
  float* S  = ws + 2 * kB * kL * kHid;      // [32][32] f32 partial sums
  float* out = (float*)d_out;

  k_zero <<<1, kB * kHid, 0, stream>>>(S);
  k_embed<<<(kB * kL * kHid) / 256, 256, 0, stream>>>(x, W0, b0, Ap, Bm);
  k_pairs<<<kB * 16, 256, 0, stream>>>(Ap, Bm, W1, b1, W2, b2, S);
  k_head <<<1, dim3(kHid, kB), 0, stream>>>(S, Wp, bp, Wo, bo, out);
}
